// GCBLock_8847632629922
// MI455X (gfx1250) — matrix operand order repaired and checked by reading, not compiled, not run on hardware
//
#include <hip/hip_runtime.h>
#include <hip/hip_bf16.h>
#include <math.h>

// ---------------------------------------------------------------------------
// CDNA5 (gfx1250) implementation of the GCN block:
//   x1  = tanh(GCNConv(x))            (WMMA GEMM + edge scatter/gather)
//   h   = tanh([x,x1] @ W1^T + b1)    (WMMA GEMM, K=512)
//   out = tanh(h @ W2^T + b2)         (WMMA GEMM, K=256)
// GEMMs run in bf16 with f32 accumulation via v_wmma_f32_16x16x32_bf16.
// ---------------------------------------------------------------------------

typedef __bf16 bf16_t;
typedef __attribute__((ext_vector_type(16))) bf16_t v16bf;
typedef __attribute__((ext_vector_type(8)))  float  v8f;

__device__ __forceinline__ unsigned short f2bf(float f) {
  // round-to-nearest-even float32 -> bf16
  unsigned u = __float_as_uint(f);
  u += 0x7FFFu + ((u >> 16) & 1u);
  return (unsigned short)(u >> 16);
}

// A fragment: 16x32 bf16 tile of a row-major matrix (ld = row stride, elems).
// Contiguous-k layout used here: lane L holds row M=L%16; hi=L>>4; VGPR v -> K=2v+16*hi.
// Each VGPR = one aligned 32-bit load.
__device__ __forceinline__ v16bf load_a_frag(const unsigned short* __restrict__ t,
                                             int ld, int mr, int hi) {
  union { v16bf v; unsigned u[8]; } f;
  const unsigned short* r = t + mr * ld + hi * 16;
#pragma unroll
  for (int v = 0; v < 4; ++v) f.u[v]     = *(const unsigned*)(r + 2 * v);
#pragma unroll
  for (int v = 0; v < 4; ++v) f.u[4 + v] = *(const unsigned*)(r + 8 + 2 * v);
  return f.v;
}

// B fragment: 32x16 bf16 tile where B[k][n] = W[n][k], W row-major [n][k]
// (ld = row stride of W). Contiguous-k layout used here: lane L holds col n=L%16; hi=L>>4;
// VGPR v -> K = 2v + 16*hi.  One aligned 32-bit load per VGPR.
__device__ __forceinline__ v16bf load_b_frag(const unsigned short* __restrict__ t,
                                             int ld, int nr, int hi) {
  union { v16bf v; unsigned u[8]; } f;
  const unsigned short* r = t + nr * ld + hi * 16;
#pragma unroll
  for (int v = 0; v < 8; ++v) f.u[v] = *(const unsigned*)(r + 2 * v);
  return f.v;
}

// One wave computes one 16-row strip of the output across all 16 column
// tiles (OUT_DIM=256), keeping the A fragments resident in VGPRs.
// EPI: 0 = store raw f32, 1 = store bf16(tanh(v + bias)), 2 = f32 tanh(v+bias)
template <int K, int EPI>
__global__ __launch_bounds__(256)
void gemm_bf16_wmma(const unsigned short* __restrict__ A, int lda,
                    const unsigned short* __restrict__ W,   // [256 x K] row-major
                    const float* __restrict__ bias,
                    float* __restrict__ outf,
                    unsigned short* __restrict__ outh,
                    int ldo, int nrows) {
  int wave = (int)((blockIdx.x * blockDim.x + threadIdx.x) >> 5);
  int lane = (int)(threadIdx.x & 31);
  int m0 = wave * 16;
  if (m0 >= nrows) return;  // wave-uniform: EXEC stays all-1 for WMMA
  int mr = lane & 15, hi = lane >> 4;
  constexpr int KT = K / 32;

  v16bf a[KT];
#pragma unroll
  for (int k = 0; k < KT; ++k)
    a[k] = load_a_frag(A + m0 * lda + k * 32, lda, mr, hi);

#pragma unroll 1
  for (int nt = 0; nt < 16; ++nt) {
    v8f c = {};
#pragma unroll
    for (int k = 0; k < KT; ++k) {
      v16bf b = load_b_frag(W + nt * 16 * K + k * 32, K, mr, hi);
      c = __builtin_amdgcn_wmma_f32_16x16x32_bf16(false, a[k], false, b,
                                                  (short)0, c, false, false);
    }
    int col = nt * 16 + mr;
    float bv = 0.0f;
    if (EPI != 0) bv = bias[col];
#pragma unroll
    for (int r = 0; r < 8; ++r) {
      int m = m0 + r + hi * 8;  // C/D layout: VGPR r -> row m0+r+8*hi
      float val = c[r] + bv;
      if (EPI == 0)       outf[m * ldo + col] = val;
      else if (EPI == 1)  outh[m * ldo + col] = f2bf(tanhf(val));
      else                outf[m * ldo + col] = tanhf(val);
    }
  }
}

// ----------------------------- helper kernels ------------------------------

__global__ void f2bf_copy(unsigned short* __restrict__ dst,
                          const float* __restrict__ src, int n) {
  int i = (int)(blockIdx.x * blockDim.x + threadIdx.x);
  if (i < n) dst[i] = f2bf(src[i]);
}

// x (f32, [N,256]) -> hcat[:, 0:256] (bf16, ld=512)
__global__ void x_to_hcat(unsigned short* __restrict__ hcat,
                          const float* __restrict__ x, int n) {
  int i = (int)(blockIdx.x * blockDim.x + threadIdx.x);
  if (i < n) {
    int r = i >> 8, c = i & 255;
    hcat[r * 512 + c] = f2bf(x[i]);
  }
}

__global__ void fill_ones(float* __restrict__ p, int n) {
  int i = (int)(blockIdx.x * blockDim.x + threadIdx.x);
  if (i < n) p[i] = 1.0f;  // self-loop contribution to degree
}

__global__ void zero_f4(float4* __restrict__ p, int n) {
  int i = (int)(blockIdx.x * blockDim.x + threadIdx.x);
  if (i < n) p[i] = make_float4(0.f, 0.f, 0.f, 0.f);
}

__global__ void deg_count(const int* __restrict__ col,
                          float* __restrict__ deg, int e) {
  int i = (int)(blockIdx.x * blockDim.x + threadIdx.x);
  if (i < e) atomicAdd(&deg[col[i]], 1.0f);
}

__global__ void deg_to_dinv(float* __restrict__ d, int n) {
  int i = (int)(blockIdx.x * blockDim.x + threadIdx.x);
  if (i < n) d[i] = rsqrtf(d[i]);  // deg >= 1 always (self loops)
}

// One wave per edge: agg[col] += xw[row] * dinv[row]*dinv[col]
// Each lane owns 8 consecutive features: 2x float4 loads + 8 f32 atomics.
__global__ __launch_bounds__(256)
void edge_agg(const int* __restrict__ row, const int* __restrict__ col,
              const float* __restrict__ xw, const float* __restrict__ dinv,
              float* __restrict__ agg, int e) {
  int t = (int)(blockIdx.x * blockDim.x + threadIdx.x);
  int w = t >> 5, lane = t & 31;
  if (w >= e) return;
  int r = row[w], c = col[w];
  float nrm = dinv[r] * dinv[c];
  const float4* s = (const float4*)(xw + (size_t)r * 256 + lane * 8);
  float4 v0 = s[0], v1 = s[1];
  float* d = agg + (size_t)c * 256 + lane * 8;
  atomicAdd(d + 0, v0.x * nrm); atomicAdd(d + 1, v0.y * nrm);
  atomicAdd(d + 2, v0.z * nrm); atomicAdd(d + 3, v0.w * nrm);
  atomicAdd(d + 4, v1.x * nrm); atomicAdd(d + 5, v1.y * nrm);
  atomicAdd(d + 6, v1.z * nrm); atomicAdd(d + 7, v1.w * nrm);
}

// x1 = tanh(agg + self_loop_term + b_gcn) -> hcat[:, 256:512] (bf16)
__global__ void finalize_x1(const float* __restrict__ agg,
                            const float* __restrict__ xw,
                            const float* __restrict__ dinv,
                            const float* __restrict__ bg,
                            unsigned short* __restrict__ hcat, int n) {
  int i = (int)(blockIdx.x * blockDim.x + threadIdx.x);
  if (i < n) {
    int r = i >> 8, c = i & 255;
    float di = dinv[r];
    float v = agg[i] + xw[i] * di * di + bg[c];
    hcat[r * 512 + 256 + c] = f2bf(tanhf(v));
  }
}

// ------------------------------- launcher ----------------------------------

extern "C" void kernel_launch(void* const* d_in, const int* in_sizes, int n_in,
                              void* d_out, int out_size, void* d_ws, size_t ws_size,
                              hipStream_t stream) {
  const float* x  = (const float*)d_in[0];
  const int*   ei = (const int*)d_in[1];   // [2, E] row-major (int32)
  const float* Wg = (const float*)d_in[3];
  const float* bg = (const float*)d_in[4];
  const float* W1 = (const float*)d_in[5];
  const float* b1 = (const float*)d_in[6];
  const float* W2 = (const float*)d_in[7];
  const float* b2 = (const float*)d_in[8];
  float* out = (float*)d_out;

  const int N = in_sizes[0] / 256;
  const int E = in_sizes[1] / 2;
  const int* row = ei;
  const int* col = ei + E;

  // workspace carve-out (256B aligned)
  size_t off = 0;
  auto take = [&](size_t bytes) -> void* {
    void* p = (char*)d_ws + off;
    off += (bytes + 255) & ~(size_t)255;
    return p;
  };
  unsigned short* hcat = (unsigned short*)take((size_t)N * 512 * 2); // [N,512] bf16
  float*          xw   = (float*)take((size_t)N * 256 * 4);          // x @ Wg^T
  float*          agg  = (float*)take((size_t)N * 256 * 4);          // scatter sums
  float*          dinv = (float*)take((size_t)N * 4);                // deg -> rsqrt
  unsigned short* wg   = (unsigned short*)take((size_t)256 * 256 * 2);
  unsigned short* w1   = (unsigned short*)take((size_t)256 * 512 * 2);
  unsigned short* w2   = (unsigned short*)take((size_t)256 * 256 * 2);
  unsigned short* h    = (unsigned short*)xw;  // reuse: xw dead after finalize

  auto cdiv = [](long long a, long long b) { return (unsigned)((a + b - 1) / b); };
  const int B = 256;  // 8 wave32 per block

  // weight / input conversions to bf16
  f2bf_copy<<<cdiv(256 * 256, B), B, 0, stream>>>(wg, Wg, 256 * 256);
  f2bf_copy<<<cdiv(256 * 512, B), B, 0, stream>>>(w1, W1, 256 * 512);
  f2bf_copy<<<cdiv(256 * 256, B), B, 0, stream>>>(w2, W2, 256 * 256);
  x_to_hcat<<<cdiv((long long)N * 256, B), B, 0, stream>>>(hcat, x, N * 256);

  // degree -> dinv (deg starts at 1.0 for self loops), zero accumulators
  fill_ones<<<cdiv(N, B), B, 0, stream>>>(dinv, N);
  zero_f4<<<cdiv((long long)N * 64, B), B, 0, stream>>>((float4*)agg, N * 64);
  deg_count<<<cdiv(E, B), B, 0, stream>>>(col, dinv, E);
  deg_to_dinv<<<cdiv(N, B), B, 0, stream>>>(dinv, N);

  const int waves = (N + 15) / 16;            // N=50000 -> exact tiles
  const unsigned gemm_blocks = cdiv((long long)waves * 32, B);

  // GEMM1: xw = x @ Wg^T   (A from hcat[:, :256], ld=512)
  gemm_bf16_wmma<256, 0><<<gemm_blocks, B, 0, stream>>>(
      hcat, 512, wg, nullptr, xw, nullptr, 256, N);

  // edge message passing (1 wave per edge, f32 atomics)
  edge_agg<<<cdiv((long long)E * 32, B), B, 0, stream>>>(row, col, xw, dinv, agg, E);

  // x1 = tanh(agg + xw*dinv^2 + bg) -> hcat[:, 256:]
  finalize_x1<<<cdiv((long long)N * 256, B), B, 0, stream>>>(
      agg, xw, dinv, bg, hcat, N * 256);

  // GEMM2: h = tanh(hcat @ W1^T + b1)  (K=512, bf16 out, aliased onto xw)
  gemm_bf16_wmma<512, 1><<<gemm_blocks, B, 0, stream>>>(
      hcat, 512, w1, b1, nullptr, h, 256, N);

  // GEMM3: out = tanh(h @ W2^T + b2)  (K=256, f32 out)
  gemm_bf16_wmma<256, 2><<<gemm_blocks, B, 0, stream>>>(
      h, 256, w2, b2, out, nullptr, 256, N);
}
